// GraphTransformerLayer_52097953300855
// MI455X (gfx1250) — compile-verified, matrix operand order checked by reading, not run on hardware
//
#include <hip/hip_runtime.h>
#include <hip/hip_bf16.h>

typedef __attribute__((ext_vector_type(16))) _Float16 v16h;
typedef __attribute__((ext_vector_type(8)))  _Float16 v8h;
typedef __attribute__((ext_vector_type(8)))  float    v8f;

#define DD 128
#define HH 8

// ---------------------------------------------------------------------------
// f32 -> f16 conversion, 8 elements/thread, vectorized.
// ---------------------------------------------------------------------------
__global__ void f32_to_f16_kernel(const float* __restrict__ X,
                                  _Float16* __restrict__ Y, long n)
{
    const long i = ((long)blockIdx.x * blockDim.x + threadIdx.x) * 8;
    if (i >= n) return;
    const float4 a = *(const float4*)(X + i);
    const float4 b = *(const float4*)(X + i + 4);
    v8h o;
    o[0] = (_Float16)a.x; o[1] = (_Float16)a.y;
    o[2] = (_Float16)a.z; o[3] = (_Float16)a.w;
    o[4] = (_Float16)b.x; o[5] = (_Float16)b.y;
    o[6] = (_Float16)b.z; o[7] = (_Float16)b.w;
    *(v8h*)(Y + i) = o;
}

// ---------------------------------------------------------------------------
// WMMA GEMM, f16 inputs / f32 accumulate, no LDS:
//   C[M,Nout] = act( A[M,K] @ W[Nout,K]^T + bias [+ R] )
// A,W pre-converted to f16. Fragments loaded straight from global memory
// (activations are L2-resident on MI455X: 192MB L2 >> 102MB max operand).
// block = 128 threads (4 wave32s); wave w -> rows [bx*64+16w,+16),
// cols [by*32,+32) as two 16x16 WMMA tiles sharing the A fragment.
// K multiple of 32, Nout multiple of 32.
// ---------------------------------------------------------------------------
__global__ void wmma_gemm_kernel(const _Float16* __restrict__ A,
                                 const _Float16* __restrict__ Wt, // [Nout,K]
                                 const float* __restrict__ bias,  // nullable
                                 const float* __restrict__ R,     // nullable
                                 float* __restrict__ C,
                                 int M, int K, int Nout, int relu)
{
    const int lane = threadIdx.x & 31;
    const int wave = threadIdx.x >> 5;
    const int m0 = blockIdx.x * 64 + wave * 16;
    const int n0 = blockIdx.y * 32;
    const int r  = lane & 15;
    const int hf = lane >> 4;

    // clamp: out-of-range A rows read row M-1; their products land only in
    // C rows >= M which are never stored (WMMA generates no exceptions).
    const int mA = (m0 + r < M) ? (m0 + r) : (M - 1);

    const _Float16* arow  = A  + (size_t)mA * K;
    const _Float16* brow0 = Wt + (size_t)(n0 + r) * K;
    const _Float16* brow1 = Wt + (size_t)(n0 + 16 + r) * K;

    v8f acc0 = {0.f, 0.f, 0.f, 0.f, 0.f, 0.f, 0.f, 0.f};
    v8f acc1 = {0.f, 0.f, 0.f, 0.f, 0.f, 0.f, 0.f, 0.f};

    for (int k0 = 0; k0 < K; k0 += 32) {
        // CDNA5 16-bit fragment layout: lane needs K = [hf*8..+7] and
        // [16+hf*8..+7] of its row -> two contiguous 16B loads each.
        const v8h alo  = *(const v8h*)(arow  + k0 + hf * 8);
        const v8h ahi  = *(const v8h*)(arow  + k0 + 16 + hf * 8);
        const v8h b0lo = *(const v8h*)(brow0 + k0 + hf * 8);
        const v8h b0hi = *(const v8h*)(brow0 + k0 + 16 + hf * 8);
        const v8h b1lo = *(const v8h*)(brow1 + k0 + hf * 8);
        const v8h b1hi = *(const v8h*)(brow1 + k0 + 16 + hf * 8);
        v16h a, b0, b1;
#pragma unroll
        for (int j = 0; j < 8; ++j) {
            a[j] = alo[j];  a[j + 8] = ahi[j];
            b0[j] = b0lo[j]; b0[j + 8] = b0hi[j];
            b1[j] = b1lo[j]; b1[j + 8] = b1hi[j];
        }
        acc0 = __builtin_amdgcn_wmma_f32_16x16x32_f16(false, a, false, b0,
                                                      (short)0, acc0, false, false);
        acc1 = __builtin_amdgcn_wmma_f32_16x16x32_f16(false, a, false, b1,
                                                      (short)0, acc1, false, false);
    }

    // C/D layout: lane holds col (lane&15); VGPR rr = row m0 + rr + 8*(lane>=16)
    const int nc0 = n0 + r, nc1 = n0 + 16 + r;
    const int mbase = m0 + (hf ? 8 : 0);
    const float bv0 = bias ? bias[nc0] : 0.f;
    const float bv1 = bias ? bias[nc1] : 0.f;
#pragma unroll
    for (int rr = 0; rr < 8; ++rr) {
        const int m = mbase + rr;
        if (m < M) {
            float v0 = acc0[rr] + bv0;
            float v1 = acc1[rr] + bv1;
            if (R) {
                v0 += R[(size_t)m * Nout + nc0];
                v1 += R[(size_t)m * Nout + nc1];
            }
            if (relu) { v0 = fmaxf(v0, 0.f); v1 = fmaxf(v1, 0.f); }
            C[(size_t)m * Nout + nc0] = v0;
            C[(size_t)m * Nout + nc1] = v1;
        }
    }
}

// ---------------------------------------------------------------------------
// Row LayerNorm over D=128. One wave per row (4 floats/lane, shfl reductions).
// ---------------------------------------------------------------------------
__global__ void ln_rows_kernel(const float* __restrict__ X,
                               const float* __restrict__ g,
                               const float* __restrict__ b,
                               float* __restrict__ Y, int Nr)
{
    const int wave = threadIdx.x >> 5, lane = threadIdx.x & 31;
    const int row = blockIdx.x * (blockDim.x >> 5) + wave;
    if (row >= Nr) return;

    const float4 v = ((const float4*)(X + (size_t)row * DD))[lane];
    float s = v.x + v.y + v.z + v.w;
#pragma unroll
    for (int off = 16; off > 0; off >>= 1) s += __shfl_xor(s, off);
    const float mean = s * (1.f / DD);

    float q = (v.x - mean) * (v.x - mean) + (v.y - mean) * (v.y - mean) +
              (v.z - mean) * (v.z - mean) + (v.w - mean) * (v.w - mean);
#pragma unroll
    for (int off = 16; off > 0; off >>= 1) q += __shfl_xor(q, off);
    const float inv = rsqrtf(q * (1.f / DD) + 1e-5f);

    const float4 gg = ((const float4*)g)[lane];
    const float4 bb = ((const float4*)b)[lane];
    float4 o;
    o.x = (v.x - mean) * inv * gg.x + bb.x;
    o.y = (v.y - mean) * inv * gg.y + bb.y;
    o.z = (v.z - mean) * inv * gg.z + bb.z;
    o.w = (v.w - mean) * inv * gg.w + bb.w;
    ((float4*)(Y + (size_t)row * DD))[lane] = o;
}

// ---------------------------------------------------------------------------
// Edge attention: per-edge dot products + exp + atomic segment sums.
// ---------------------------------------------------------------------------
__global__ void edge_attn_kernel(const float* __restrict__ Q,
                                 const float* __restrict__ Km,
                                 const float* __restrict__ Vm,
                                 const int* __restrict__ src,
                                 const int* __restrict__ dst,
                                 float* __restrict__ wV,
                                 float* __restrict__ z, int E)
{
    const int e = blockIdx.x * blockDim.x + threadIdx.x;
    if (e >= E) return;
    const int s = src[e], d = dst[e];
    const float4* kq = (const float4*)(Km + (size_t)s * DD);
    const float4* qq = (const float4*)(Q  + (size_t)d * DD);
    const float4* vq = (const float4*)(Vm + (size_t)s * DD);
#pragma unroll
    for (int h = 0; h < HH; ++h) {
        float sc = 0.f;
#pragma unroll
        for (int j = 0; j < 4; ++j) {
            const float4 k4 = kq[h * 4 + j], q4 = qq[h * 4 + j];
            sc += k4.x * q4.x + k4.y * q4.y + k4.z * q4.z + k4.w * q4.w;
        }
        sc = fminf(fmaxf(sc * 0.25f, -5.f), 5.f); // 1/sqrt(16)=0.25, clip
        sc = __expf(sc);
#pragma unroll
        for (int j = 0; j < 4; ++j) {
            const float4 v4 = vq[h * 4 + j];
            float* base = wV + (size_t)d * DD + h * 16 + j * 4;
            atomicAdd(base + 0, v4.x * sc);
            atomicAdd(base + 1, v4.y * sc);
            atomicAdd(base + 2, v4.z * sc);
            atomicAdd(base + 3, v4.w * sc);
        }
        atomicAdd(z + (size_t)d * HH + h, sc);
    }
}

// y = hsub + wV / z (broadcast z per head)
__global__ void add_attn_kernel(const float* __restrict__ hsub,
                                const float* __restrict__ wV,
                                const float* __restrict__ z,
                                float* __restrict__ Y, int Nr)
{
    const long idx = (long)blockIdx.x * blockDim.x + threadIdx.x;
    if (idx >= (long)Nr * DD) return;
    const int row = (int)(idx >> 7), c = (int)(idx & 127), h = c >> 4;
    Y[idx] = hsub[idx] + wV[idx] / z[(size_t)row * HH + h];
}

// BatchNorm statistics: per-column sum & sumsq via grid-stride + atomics.
__global__ void bn_stats_kernel(const float* __restrict__ X,
                                float* __restrict__ sums,
                                float* __restrict__ sqs, int Nr)
{
    const int c = threadIdx.x; // 128 threads, one column each
    float s = 0.f, q = 0.f;
    for (int rr = blockIdx.x; rr < Nr; rr += gridDim.x) {
        const float v = X[(size_t)rr * DD + c];
        s += v; q += v * v;
    }
    atomicAdd(&sums[c], s);
    atomicAdd(&sqs[c], q);
}

__global__ void bn_apply_kernel(const float* __restrict__ X,
                                const float* __restrict__ sums,
                                const float* __restrict__ sqs,
                                const float* __restrict__ g,
                                const float* __restrict__ b,
                                float* __restrict__ Y, int Nr)
{
    const long idx = (long)blockIdx.x * blockDim.x + threadIdx.x;
    if (idx >= (long)Nr * DD) return;
    const int c = (int)(idx & 127);
    const float invN = 1.f / (float)Nr;
    const float mean = sums[c] * invN;
    const float var  = sqs[c] * invN - mean * mean;
    Y[idx] = (X[idx] - mean) * rsqrtf(var + 1e-5f) * g[c] + b[c];
}

// ---------------------------------------------------------------------------
extern "C" void kernel_launch(void* const* d_in, const int* in_sizes, int n_in,
                              void* d_out, int out_size, void* d_ws, size_t ws_size,
                              hipStream_t stream)
{
    const float* h    = (const float*)d_in[0];
    const int*   src  = (const int*)  d_in[1];
    const int*   dst  = (const int*)  d_in[2];
    const float* Win  = (const float*)d_in[3];
    const float* bin_ = (const float*)d_in[4];
    const float* Wout = (const float*)d_in[5];
    const float* bout = (const float*)d_in[6];
    const float* Wte1 = (const float*)d_in[7];
    const float* bte1 = (const float*)d_in[8];
    const float* Wte2 = (const float*)d_in[9];
    const float* bte2 = (const float*)d_in[10];
    const float* ln1g = (const float*)d_in[11];
    const float* ln1b = (const float*)d_in[12];
    const float* ln2g = (const float*)d_in[13];
    const float* ln2b = (const float*)d_in[14];
    const float* WQ   = (const float*)d_in[15];
    const float* WK   = (const float*)d_in[16];
    const float* WV   = (const float*)d_in[17];
    const float* W1   = (const float*)d_in[18];
    const float* b1   = (const float*)d_in[19];
    const float* W2   = (const float*)d_in[20];
    const float* b2   = (const float*)d_in[21];
    const float* bn1g = (const float*)d_in[22];
    const float* bn1b = (const float*)d_in[23];
    const float* bn2g = (const float*)d_in[24];
    const float* bn2b = (const float*)d_in[25];

    const int N = in_sizes[0] / DD;   // 50000
    const int E = in_sizes[1];        // 500000
    const size_t ND = (size_t)N * DD;

    // -------- workspace layout (floats) --------
    float* R0 = (float*)d_ws;           // 6 activation regions of N*D floats
    float* R1 = R0 + ND;
    float* R2 = R1 + ND;
    float* R3 = R2 + ND;
    float* R4 = R3 + ND;
    /* R5 implicit = R4 + ND */
    float* zb    = R0 + 6 * ND;         // [N, H]
    float* stats = zb + (size_t)N * HH; // 512 floats
    _Float16* HA = (_Float16*)(stats + 512);      // activations f16, up to 4*ND halves
    _Float16* HW = HA + 4 * ND;                   // weights f16 pool
    // weight pool sub-offsets (halves)
    _Float16* hWinV = HW;                 // 128x128
    _Float16* hWout = HW + 16384;         // 128x128
    _Float16* hWte1 = HW + 32768;         // 512x128
    _Float16* hWte2 = HW + 98304;         // 128x512
    _Float16* hWQ   = HW + 163840;        // 128x128
    _Float16* hWK   = HW + 180224;
    _Float16* hWV   = HW + 196608;
    _Float16* hW1   = HW + 212992;        // 256x128
    _Float16* hW2   = HW + 245760;        // 128x256

    auto cvt = [&](const float* X, _Float16* Y, long n) {
        f32_to_f16_kernel<<<(int)((n / 8 + 255) / 256), 256, 0, stream>>>(X, Y, n);
    };
    auto gemm = [&](const _Float16* A, const _Float16* Wt, const float* bias,
                    const float* R, float* C, int K, int Nout, int relu) {
        dim3 grid((N + 63) / 64, Nout / 32);
        wmma_gemm_kernel<<<grid, 128, 0, stream>>>(A, Wt, bias, R, C, N, K, Nout, relu);
    };
    const int ELEM_BLOCKS = (int)((ND + 255) / 256);
    const int LN_BLOCKS   = (N + 7) / 8;

    // -------- prologue: zero accumulators, convert weights once --------
    hipMemsetAsync(stats, 0, 512 * sizeof(float), stream);
    cvt(Win + 2 * DD * DD, hWinV, DD * DD);
    cvt(Wout, hWout, DD * DD);
    cvt(Wte1, hWte1, 4 * DD * DD);
    cvt(Wte2, hWte2, 4 * DD * DD);
    cvt(WQ, hWQ, DD * DD);
    cvt(WK, hWK, DD * DD);
    cvt(WV, hWV, DD * DD);
    cvt(W1, hW1, 2 * DD * DD);
    cvt(W2, hW2, 2 * DD * DD);

    // 1) v = h @ WinV^T + binV
    cvt(h, HA, (long)ND);
    gemm(HA, hWinV, bin_ + 2 * DD, nullptr, R0, DD, DD, 0);
    // 2) pre1 = h + v @ Wout^T + bout
    cvt(R0, HA, (long)ND);
    gemm(HA, hWout, bout, h, R1, DD, DD, 0);
    // 3) x = LN1(pre1) -> R0
    ln_rows_kernel<<<LN_BLOCKS, 256, 0, stream>>>(R1, ln1g, ln1b, R0, N);
    // 4) ff1 = relu(x @ Wte1^T + bte1)    [N,512] in R2..R5
    cvt(R0, HA, (long)ND);
    gemm(HA, hWte1, bte1, nullptr, R2, DD, 4 * DD, 1);
    // 5) pre2 = x + ff1 @ Wte2^T + bte2 -> R1
    cvt(R2, HA, (long)(4 * ND));
    gemm(HA, hWte2, bte2, R0, R1, 4 * DD, DD, 0);
    // 6) hsub = LN2(pre2) -> R2
    ln_rows_kernel<<<LN_BLOCKS, 256, 0, stream>>>(R1, ln2g, ln2b, R2, N);
    // 7) Q/K/V projections (no bias), shared f16 A
    cvt(R2, HA, (long)ND);
    gemm(HA, hWQ, nullptr, nullptr, R0, DD, DD, 0);
    gemm(HA, hWK, nullptr, nullptr, R1, DD, DD, 0);
    gemm(HA, hWV, nullptr, nullptr, R3, DD, DD, 0);
    // 8) segment-sum attention (atomics) into wV=R4, z=zb
    hipMemsetAsync(R4, 0, ND * sizeof(float), stream);
    hipMemsetAsync(zb, 0, (size_t)N * HH * sizeof(float), stream);
    edge_attn_kernel<<<(E + 255) / 256, 256, 0, stream>>>(R0, R1, R3, src, dst, R4, zb, E);
    // 9) y = hsub + wV / z -> R0
    add_attn_kernel<<<ELEM_BLOCKS, 256, 0, stream>>>(R2, R4, zb, R0, N);
    // 10) hh = BN1(y) -> R1
    bn_stats_kernel<<<1024, DD, 0, stream>>>(R0, stats, stats + 128, N);
    bn_apply_kernel<<<ELEM_BLOCKS, 256, 0, stream>>>(R0, stats, stats + 128, bn1g, bn1b, R1, N);
    // 11) f2a = relu(hh @ W1^T + b1)      [N,256] in R3..R4
    cvt(R1, HA, (long)ND);
    gemm(HA, hW1, b1, nullptr, R3, DD, 2 * DD, 1);
    // 12) y2 = hh + f2a @ W2^T + b2 -> R0
    cvt(R3, HA, (long)(2 * ND));
    gemm(HA, hW2, b2, R1, R0, 2 * DD, DD, 0);
    // 13) out = BN2(y2)
    bn_stats_kernel<<<1024, DD, 0, stream>>>(R0, stats + 256, stats + 384, N);
    bn_apply_kernel<<<ELEM_BLOCKS, 256, 0, stream>>>(R0, stats + 256, stats + 384,
                                                     bn2g, bn2b, (float*)d_out, N);
}
